// Attention_24764781429249
// MI455X (gfx1250) — matrix operand order repaired and checked by reading, not compiled, not run on hardware
//
#include <hip/hip_runtime.h>
#include <hip/hip_bf16.h>

typedef _Float16 h16 __attribute__((ext_vector_type(16)));
typedef _Float16 h8v __attribute__((ext_vector_type(8)));
typedef float    f8v __attribute__((ext_vector_type(8)));

union H16u { h16 v; h8v h[2]; };

#define WMMA_F16(a, b, c) \
  __builtin_amdgcn_wmma_f32_16x16x32_f16(false, (a), false, (b), (short)0, (c), false, false)

// A-fragment (16x32 f16, M x K): lane l holds row M=l%16.
// j=0..7  -> K = (l/16)*8 + j ; j=8..15 -> K = 16 + (l/16)*8 + (j-8)
__device__ __forceinline__ h16 load_frag_a(const _Float16* __restrict__ base,
                                           int row, int ld, int colbase, int lane) {
  const _Float16* p = base + (size_t)row * ld + colbase + ((lane >> 4) << 3);
  H16u u;
  u.h[0] = *(const h8v*)p;
  u.h[1] = *(const h8v*)(p + 16);
  return u.v;
}

// B-fragment (32x16 f16, K x N): lane l holds column N=l%16.
// j=0..7 -> K = (l/16)*8 + j ; j=8..15 -> K = 16 + (l/16)*8 + (j-8). 'row' is the memory row
// read contiguously (W row / K row / Vt row / LDS tile row).
__device__ __forceinline__ h16 load_frag_b(const _Float16* __restrict__ base,
                                           int row, int ld, int colbase, int lane) {
  const _Float16* p = base + (size_t)row * ld + colbase + ((lane >> 4) << 3);
  H16u u;
  u.h[0] = *(const h8v*)p;
  u.h[1] = *(const h8v*)(p + 16);
  return u.v;
}

// gfx1250 async copy: global -> LDS, tracked by ASYNCcnt (no VGPR round-trip).
__device__ __forceinline__ void async_g2l_b128(const void* gptr, const void* lptr) {
  unsigned laddr = (unsigned)(uintptr_t)lptr;            // low 32 bits = LDS byte addr
  unsigned long long gaddr = (unsigned long long)(uintptr_t)gptr;
  asm volatile("global_load_async_to_lds_b128 %0, %1, off"
               :: "v"(laddr), "v"(gaddr) : "memory");
}

// ---------------------------------------------------------------- f32 -> f16
__global__ void cvt_f32_f16(const float* __restrict__ in, _Float16* __restrict__ out, int n) {
  int i = blockIdx.x * blockDim.x + threadIdx.x;
  if (i < n) out[i] = (_Float16)in[i];
}

// ---------------------------------------------------------------- QKV GEMM
// Y[m][n] = sum_k xh[m][k] * W[n][k]  (x @ W^T). Wave tile 32x64 -> 8 WMMAs
// per K-step for 12 b128 loads (1.5 loads/WMMA).
__global__ __launch_bounds__(128) void gemm_qkv(
    const _Float16* __restrict__ xh,
    const _Float16* __restrict__ wq, const _Float16* __restrict__ wk,
    const _Float16* __restrict__ wv,
    _Float16* __restrict__ Qh, _Float16* __restrict__ Kh, _Float16* __restrict__ Vt) {
  const int lane  = threadIdx.x & 31;
  const int wave  = threadIdx.x >> 5;
  const int which = blockIdx.z;
  const _Float16* W = (which == 0) ? wq : ((which == 1) ? wk : wv);
  const int m0 = blockIdx.x * 128 + wave * 32;  // gridDim.x = 4096/128 = 32
  const int n0 = blockIdx.y * 64;               // gridDim.y = 1024/64 = 16

  f8v acc[8] = {};   // [mi*4 + nn]
  for (int kb = 0; kb < 1024; kb += 32) {
    h16 a0 = load_frag_a(xh, m0 +      (lane & 15), 1024, kb, lane);
    h16 a1 = load_frag_a(xh, m0 + 16 + (lane & 15), 1024, kb, lane);
#pragma unroll
    for (int nn = 0; nn < 4; ++nn) {
      h16 b = load_frag_b(W, n0 + nn * 16 + (lane & 15), 1024, kb, lane);
      acc[nn]     = WMMA_F16(a0, b, acc[nn]);
      acc[4 + nn] = WMMA_F16(a1, b, acc[4 + nn]);
    }
  }

  if (which < 2) {
    _Float16* out = (which == 0) ? Qh : Kh;
#pragma unroll
    for (int mi = 0; mi < 2; ++mi)
#pragma unroll
      for (int nn = 0; nn < 4; ++nn)
#pragma unroll
        for (int r = 0; r < 8; ++r) {
          int M = m0 + mi * 16 + r + ((lane >> 4) << 3);
          int n = n0 + nn * 16 + (lane & 15);
          out[(size_t)M * 1024 + n] = (_Float16)acc[mi * 4 + nn][r];
        }
  } else {
#pragma unroll
    for (int mi = 0; mi < 2; ++mi)
#pragma unroll
      for (int nn = 0; nn < 4; ++nn)
#pragma unroll
        for (int r = 0; r < 8; ++r) {
          int M = m0 + mi * 16 + r + ((lane >> 4) << 3);
          int n = n0 + nn * 16 + (lane & 15);
          int b = M >> 11, t = M & 2047;   // batch, token
          int h = n >> 6,  d = n & 63;     // head, head-dim
          Vt[((size_t)((b * 16 + h) * 64 + d) << 11) + t] = (_Float16)acc[mi * 4 + nn][r];
        }
  }
}

// ---------------------------------------------------------------- attention
// One wave per 16-query tile; 4 waves/block share (b,h) and the key loop, so
// K/V chunks are staged in LDS once per block via async-to-LDS.
#define KT_STRIDE 72   // halfs: 64 + 8 pad -> 144B rows (16B aligned, spreads banks)
#define VT_STRIDE 40   // halfs: 32 + 8 pad -> 80B rows
#define P_STRIDE  56   // halfs: 112B rows
__global__ __launch_bounds__(128) void attn(
    const _Float16* __restrict__ Qh, const _Float16* __restrict__ Kh,
    const _Float16* __restrict__ Vt, _Float16* __restrict__ Oh) {
  __shared__ __align__(16) _Float16 sK[32 * KT_STRIDE];        // 4608 B
  __shared__ __align__(16) _Float16 sV[64 * VT_STRIDE];        // 5120 B
  __shared__ __align__(16) _Float16 sP[4 * 16 * P_STRIDE];     // 7168 B

  const int lane = threadIdx.x & 31;
  const int wave = threadIdx.x >> 5;
  const int tid  = threadIdx.x;
  const int b = blockIdx.x >> 4;       // gridDim.x = B*H = 32
  const int h = blockIdx.x & 15;
  const int q0 = blockIdx.y * 64 + wave * 16;   // gridDim.y = 2048/64 = 32
  const int rowg = b * 2048 + q0 + (lane & 15);

  h16 qa0 = load_frag_a(Qh, rowg, 1024, h * 64, lane);
  h16 qa1 = load_frag_a(Qh, rowg, 1024, h * 64 + 32, lane);

  f8v o0 = {}, o1 = {}, o2 = {}, o3 = {};
  float mrow[8], lrow[8];
#pragma unroll
  for (int r = 0; r < 8; ++r) { mrow[r] = -3.0e38f; lrow[r] = 0.f; }

  _Float16* myl = sP + wave * 16 * P_STRIDE;
  const _Float16* kbase = Kh + (size_t)b * 2048 * 1024 + h * 64;
  const _Float16* vbase = Vt + ((size_t)(b * 16 + h) << 17);  // *64*2048

  for (int kc = 0; kc < 2048; kc += 32) {
    // ---- stage K tile (32 keys x 64 d) and V tile (64 d x 32 keys) in LDS
#pragma unroll
    for (int i = 0; i < 2; ++i) {
      int c = tid + i * 128;                       // 0..255: 16B chunks
      int kr = c >> 3, kcol = (c & 7) << 3;        // K tile: 8 chunks / row
      async_g2l_b128(kbase + (size_t)(kc + kr) * 1024 + kcol,
                     sK + kr * KT_STRIDE + kcol);
      int vr = c >> 2, vcol = (c & 3) << 3;        // V tile: 4 chunks / row
      async_g2l_b128(vbase + (size_t)vr * 2048 + kc + vcol,
                     sV + vr * VT_STRIDE + vcol);
    }
    asm volatile("s_wait_asynccnt 0" ::: "memory");
    __syncthreads();

    // ---- S = Q * K^T  (K-dim = d, two 32-wide steps), frags from LDS
    f8v s0 = {}, s1 = {};
    {
      h16 kb0 = load_frag_b(sK,      (lane & 15), KT_STRIDE, 0,  lane);
      h16 kb1 = load_frag_b(sK, 16 + (lane & 15), KT_STRIDE, 0,  lane);
      s0 = WMMA_F16(qa0, kb0, s0);
      s1 = WMMA_F16(qa0, kb1, s1);
      h16 kb2 = load_frag_b(sK,      (lane & 15), KT_STRIDE, 32, lane);
      h16 kb3 = load_frag_b(sK, 16 + (lane & 15), KT_STRIDE, 32, lane);
      s0 = WMMA_F16(qa1, kb2, s0);
      s1 = WMMA_F16(qa1, kb3, s1);
    }

    // ---- online softmax (16-lane reductions match the C-layout halves)
    float alpha[8];
#pragma unroll
    for (int r = 0; r < 8; ++r) {
      float v0 = s0[r] * 0.125f, v1 = s1[r] * 0.125f;  // scale = D^-0.5
      float mx = fmaxf(v0, v1);
#pragma unroll
      for (int msk = 1; msk < 16; msk <<= 1) mx = fmaxf(mx, __shfl_xor(mx, msk, 16));
      float mnew = fmaxf(mrow[r], mx);
      alpha[r] = __expf(mrow[r] - mnew);
      mrow[r] = mnew;
      float p0 = __expf(v0 - mnew), p1 = __expf(v1 - mnew);
      s0[r] = p0; s1[r] = p1;
      float rs = p0 + p1;
#pragma unroll
      for (int msk = 1; msk < 16; msk <<= 1) rs += __shfl_xor(rs, msk, 16);
      lrow[r] = lrow[r] * alpha[r] + rs;
    }
#pragma unroll
    for (int r = 0; r < 8; ++r) {
      o0[r] *= alpha[r]; o1[r] *= alpha[r]; o2[r] *= alpha[r]; o3[r] *= alpha[r];
    }

    // ---- C-layout -> A-layout for P via per-wave LDS tile (in-order DS)
#pragma unroll
    for (int r = 0; r < 8; ++r) {
      int row = r + ((lane >> 4) << 3);
      myl[row * P_STRIDE +      (lane & 15)] = (_Float16)s0[r];
      myl[row * P_STRIDE + 16 + (lane & 15)] = (_Float16)s1[r];
    }
    asm volatile("s_wait_dscnt 0" ::: "memory");
    H16u pu;
    const _Float16* pp = myl + (lane & 15) * P_STRIDE + ((lane >> 4) << 3);
    pu.h[0] = *(const h8v*)pp;
    pu.h[1] = *(const h8v*)(pp + 16);

    // ---- O += P * V  (V B-frags from LDS, contiguous along keys)
    h16 vf0 = load_frag_b(sV,      (lane & 15), VT_STRIDE, 0, lane);
    h16 vf1 = load_frag_b(sV, 16 + (lane & 15), VT_STRIDE, 0, lane);
    h16 vf2 = load_frag_b(sV, 32 + (lane & 15), VT_STRIDE, 0, lane);
    h16 vf3 = load_frag_b(sV, 48 + (lane & 15), VT_STRIDE, 0, lane);
    o0 = WMMA_F16(pu.v, vf0, o0);
    o1 = WMMA_F16(pu.v, vf1, o1);
    o2 = WMMA_F16(pu.v, vf2, o2);
    o3 = WMMA_F16(pu.v, vf3, o3);

    __syncthreads();   // before next iteration overwrites sK/sV
  }

#pragma unroll
  for (int r = 0; r < 8; ++r) {
    float inv = 1.0f / lrow[r];
    int M = b * 2048 + q0 + r + ((lane >> 4) << 3);
    size_t ro = (size_t)M * 1024 + h * 64 + (lane & 15);
    Oh[ro]      = (_Float16)(o0[r] * inv);
    Oh[ro + 16] = (_Float16)(o1[r] * inv);
    Oh[ro + 32] = (_Float16)(o2[r] * inv);
    Oh[ro + 48] = (_Float16)(o3[r] * inv);
  }
}

// ---------------------------------------------------------------- projection
__global__ __launch_bounds__(128) void gemm_proj(
    const _Float16* __restrict__ Oh, const _Float16* __restrict__ Wp,
    const float* __restrict__ bias, float* __restrict__ out) {
  const int lane = threadIdx.x & 31;
  const int wave = threadIdx.x >> 5;
  const int m0 = blockIdx.x * 128 + wave * 32;  // gridDim.x = 32
  const int n0 = blockIdx.y * 64;

  f8v acc[8] = {};
  for (int kb = 0; kb < 1024; kb += 32) {
    h16 a0 = load_frag_a(Oh, m0 +      (lane & 15), 1024, kb, lane);
    h16 a1 = load_frag_a(Oh, m0 + 16 + (lane & 15), 1024, kb, lane);
#pragma unroll
    for (int nn = 0; nn < 4; ++nn) {
      h16 b = load_frag_b(Wp, n0 + nn * 16 + (lane & 15), 1024, kb, lane);
      acc[nn]     = WMMA_F16(a0, b, acc[nn]);
      acc[4 + nn] = WMMA_F16(a1, b, acc[4 + nn]);
    }
  }
#pragma unroll
  for (int mi = 0; mi < 2; ++mi)
#pragma unroll
    for (int nn = 0; nn < 4; ++nn)
#pragma unroll
      for (int r = 0; r < 8; ++r) {
        int M = m0 + mi * 16 + r + ((lane >> 4) << 3);
        int n = n0 + nn * 16 + (lane & 15);
        out[(size_t)M * 1024 + n] = acc[mi * 4 + nn][r] + bias[n];
      }
}

// ---------------------------------------------------------------- launcher
extern "C" void kernel_launch(void* const* d_in, const int* in_sizes, int n_in,
                              void* d_out, int out_size, void* d_ws, size_t ws_size,
                              hipStream_t stream) {
  const float* x  = (const float*)d_in[0];
  const float* Wq = (const float*)d_in[1];
  const float* Wk = (const float*)d_in[2];
  const float* Wv = (const float*)d_in[3];
  const float* Wp = (const float*)d_in[4];
  const float* bp = (const float*)d_in[5];

  const int XN = 4096 * 1024;   // B*N*C
  const int WN = 1024 * 1024;   // C*C

  _Float16* ws  = (_Float16*)d_ws;
  _Float16* xh  = ws;
  _Float16* wqh = xh  + XN;
  _Float16* wkh = wqh + WN;
  _Float16* wvh = wkh + WN;
  _Float16* wph = wvh + WN;
  _Float16* Qh  = wph + WN;
  _Float16* Kh  = Qh + XN;
  _Float16* Vt  = Kh + XN;      // [B][H][D][N]
  _Float16* Oh  = Vt + XN;      // total 48 MB

  cvt_f32_f16<<<(XN + 255) / 256, 256, 0, stream>>>(x,  xh,  XN);
  cvt_f32_f16<<<(WN + 255) / 256, 256, 0, stream>>>(Wq, wqh, WN);
  cvt_f32_f16<<<(WN + 255) / 256, 256, 0, stream>>>(Wk, wkh, WN);
  cvt_f32_f16<<<(WN + 255) / 256, 256, 0, stream>>>(Wv, wvh, WN);
  cvt_f32_f16<<<(WN + 255) / 256, 256, 0, stream>>>(Wp, wph, WN);

  gemm_qkv<<<dim3(32, 16, 3), 128, 0, stream>>>(xh, wqh, wkh, wvh, Qh, Kh, Vt);
  attn<<<dim3(32, 32), 128, 0, stream>>>(Qh, Kh, Vt, Oh);
  gemm_proj<<<dim3(32, 16), 128, 0, stream>>>(Oh, wph, bp, (float*)d_out);
}
